// EncoderBlock_43370579755511
// MI455X (gfx1250) — compile-verified, matrix operand order checked by reading, not run on hardware
//
#include <hip/hip_runtime.h>

// ---------------------------------------------------------------------------
// Mamba block + downsample conv + LayerNorm for MI455X (gfx1250, wave32).
// GEMMs via v_wmma_f32_16x16x32_bf16 with pre-transposed bf16 weights,
// async global->LDS tile copies (ASYNCcnt) with LDS double-buffering;
// scan via 3-phase chunked recurrence.
// ---------------------------------------------------------------------------

#define BATCH 16
#define SEQL  2048
#define DMODEL 256
#define DINNER 512
#define NSTATE 16
#define DTRANK 16
#define NCHUNK 32
#define CHUNKL 64   // SEQL / NCHUNK

#if defined(__HIP_DEVICE_COMPILE__) &&                                        \
    __has_builtin(__builtin_amdgcn_global_load_async_to_lds_b128) &&          \
    __has_builtin(__builtin_amdgcn_s_wait_asynccnt)
#define USE_ASYNC_LDS 1
#else
#define USE_ASYNC_LDS 0
#endif

typedef __attribute__((ext_vector_type(16))) __bf16 v16bf;
typedef __attribute__((ext_vector_type(8)))  float  v8f;
typedef int v4i_t __attribute__((ext_vector_type(4)));
#if USE_ASYNC_LDS
typedef __attribute__((address_space(1))) v4i_t g_v4i;  // global int4
typedef __attribute__((address_space(3))) v4i_t l_v4i;  // LDS int4
#endif

__device__ __forceinline__ unsigned short f32_to_bf16(float f) {
  unsigned int u = __float_as_uint(f);
  unsigned int r = (u + 0x7FFFu + ((u >> 16) & 1u)) >> 16;
  return (unsigned short)r;
}

// ---------------------------------------------------------------------------
// generic f32 -> bf16 convert (row-major copy, used for activations)
// ---------------------------------------------------------------------------
__global__ void k_f32_to_bf16(const float* __restrict__ in,
                              unsigned short* __restrict__ out, int n) {
  int i = blockIdx.x * blockDim.x + threadIdx.x;
  if (i < n) out[i] = f32_to_bf16(in[i]);
}

// weight convert + transpose: in[K][N] f32 -> out[N][K] bf16
__global__ void k_cvt_transpose(const float* __restrict__ in,
                                unsigned short* __restrict__ out, int K,
                                int N) {
  int i = blockIdx.x * blockDim.x + threadIdx.x;
  if (i >= K * N) return;
  int k = i / N, n = i % N;
  out[(size_t)n * K + k] = f32_to_bf16(in[i]);
}

// down_w [O=256][I=256][3] -> bf16 Wt[o][k=tap*256+i]  (already N x K layout)
__global__ void k_wdown_cvt(const float* __restrict__ dw,
                            unsigned short* __restrict__ out) {
  int i = blockIdx.x * blockDim.x + threadIdx.x;
  if (i >= 256 * 768) return;
  int o = i / 768, kk = i % 768;
  int tap = kk / 256, ci = kk % 256;
  out[(size_t)o * 768 + kk] = f32_to_bf16(dw[((size_t)o * 256 + ci) * 3 + tap]);
}

// ---------------------------------------------------------------------------
// WMMA bf16 GEMM: C[M,N] = A[M,K] * Bt[N,K]^T, f32 accumulate.
// Block tile 128x64, K-step 64, 8 waves (wave32) in a 4x2 grid,
// each wave computes a 32x32 sub-tile = 8x v_wmma_f32_16x16x32_bf16 per step.
// Tiles staged with GLOBAL_LOAD_ASYNC_TO_LDS_B128 + double buffering.
// M must be a multiple of 128; K a multiple of 64.
// MODE 0: plain f32 store to out0 (leading dim ldo)
// MODE 1: column split at 512: cols [0,512)->out0, [512,1024)->out1 (ld 512)
// GUARD_N: bounds-check N on B-tile loads and C stores (only x_proj, N=48)
// ---------------------------------------------------------------------------
template <int MODE, bool GUARD_N>
__global__ __launch_bounds__(256) void k_wmma_gemm(
    const unsigned short* __restrict__ A, int lda,
    const unsigned short* __restrict__ Bt, int ldb,
    float* __restrict__ out0, float* __restrict__ out1, int ldo,
    int M, int N, int K) {
  __shared__ __align__(16) unsigned short sA[2][128 * 64];
  __shared__ __align__(16) unsigned short sB[2][64 * 64];  // [n][k]

  const int tid  = threadIdx.x;
  const int lane = tid & 31;
  const int w    = tid >> 5;
  const int wm   = w & 3;        // wave M position (0..3)
  const int wn   = w >> 2;       // wave N position (0..1)
  const int half = lane >> 4;    // lane half (K-half select per ISA layout)
  const int mrow = lane & 15;    // row (A) / col (B) within fragment

  const int m0 = blockIdx.x * 128;
  const int n0 = blockIdx.y * 64;

  v8f acc[2][2] = {};
  union Frag { v16bf v; uint4 q[2]; };

  // stage one 128x64 A tile + 64x64 B tile into LDS buffer bb
  auto issue_tile = [&](int k0, int bb) {
#if USE_ASYNC_LDS
    #pragma unroll
    for (int j = 0; j < 4; ++j) {
      int r = (tid >> 3) + 32 * j, q = tid & 7;
      __builtin_amdgcn_global_load_async_to_lds_b128(
          (g_v4i*)&A[(size_t)(m0 + r) * lda + k0 + q * 8],
          (l_v4i*)&sA[bb][r * 64 + q * 8], 0, 0);
    }
    #pragma unroll
    for (int j = 0; j < 2; ++j) {
      int r = (tid >> 3) + 32 * j, q = tid & 7;
      if (!GUARD_N || (n0 + r) < N) {
        __builtin_amdgcn_global_load_async_to_lds_b128(
            (g_v4i*)&Bt[(size_t)(n0 + r) * ldb + k0 + q * 8],
            (l_v4i*)&sB[bb][r * 64 + q * 8], 0, 0);
      } else {
        *(uint4*)&sB[bb][r * 64 + q * 8] = make_uint4(0u, 0u, 0u, 0u);
      }
    }
#else
    #pragma unroll
    for (int j = 0; j < 4; ++j) {
      int r = (tid >> 3) + 32 * j, q = tid & 7;
      *(uint4*)&sA[bb][r * 64 + q * 8] =
          *(const uint4*)&A[(size_t)(m0 + r) * lda + k0 + q * 8];
    }
    #pragma unroll
    for (int j = 0; j < 2; ++j) {
      int r = (tid >> 3) + 32 * j, q = tid & 7;
      uint4 val = make_uint4(0u, 0u, 0u, 0u);
      if (!GUARD_N || (n0 + r) < N)
        val = *(const uint4*)&Bt[(size_t)(n0 + r) * ldb + k0 + q * 8];
      *(uint4*)&sB[bb][r * 64 + q * 8] = val;
    }
#endif
  };

  // fragments per 16-bit WMMA layout: lane half h holds K in
  // {8h..8h+7} (VGPR 0-3) and {16+8h..16+8h+7} (VGPR 4-7) -> 2 b128 loads
  auto compute_tile = [&](int bb) {
    #pragma unroll
    for (int kk = 0; kk < 2; ++kk) {
      Frag aF[2], bF[2];
      #pragma unroll
      for (int mf = 0; mf < 2; ++mf) {
        int row = wm * 32 + mf * 16 + mrow;
        aF[mf].q[0] = *(const uint4*)&sA[bb][row * 64 + kk * 32 + 8 * half];
        aF[mf].q[1] =
            *(const uint4*)&sA[bb][row * 64 + kk * 32 + 16 + 8 * half];
      }
      #pragma unroll
      for (int nf = 0; nf < 2; ++nf) {
        int col = wn * 32 + nf * 16 + mrow;
        bF[nf].q[0] = *(const uint4*)&sB[bb][col * 64 + kk * 32 + 8 * half];
        bF[nf].q[1] =
            *(const uint4*)&sB[bb][col * 64 + kk * 32 + 16 + 8 * half];
      }
      #pragma unroll
      for (int mf = 0; mf < 2; ++mf)
        #pragma unroll
        for (int nf = 0; nf < 2; ++nf)
          acc[mf][nf] = __builtin_amdgcn_wmma_f32_16x16x32_bf16(
              false, aF[mf].v, false, bF[nf].v, (short)0, acc[mf][nf],
              false, false);
    }
  };

#if USE_ASYNC_LDS
  const int nk = K >> 6;
  issue_tile(0, 0);
  for (int it = 0; it < nk; ++it) {
    int cur = it & 1;
    __builtin_amdgcn_s_wait_asynccnt(0);
    __syncthreads();
    if (it + 1 < nk) issue_tile((it + 1) << 6, cur ^ 1);
    compute_tile(cur);
  }
#else
  for (int k0 = 0; k0 < K; k0 += 64) {
    issue_tile(k0, 0);
    __syncthreads();
    compute_tile(0);
    __syncthreads();
  }
#endif

  // --- epilogue: C/D layout: VGPR r -> M = r + 8*half, N = lane&15
  #pragma unroll
  for (int mf = 0; mf < 2; ++mf) {
    #pragma unroll
    for (int nf = 0; nf < 2; ++nf) {
      int gcol = n0 + wn * 32 + nf * 16 + (lane & 15);
      #pragma unroll
      for (int r = 0; r < 8; ++r) {
        int grow = m0 + wm * 32 + mf * 16 + half * 8 + r;
        float v = acc[mf][nf][r];
        if (MODE == 0) {
          if (!GUARD_N || gcol < N) out0[(size_t)grow * ldo + gcol] = v;
        } else {
          if (gcol < 512) out0[(size_t)grow * 512 + gcol] = v;
          else            out1[(size_t)grow * 512 + (gcol - 512)] = v;
        }
      }
    }
  }
}

// ---------------------------------------------------------------------------
// causal depthwise conv (width 4) + bias + SiLU; emits f32 and bf16 copies
// ---------------------------------------------------------------------------
__global__ void k_conv_silu(const float* __restrict__ xm_raw,
                            const float* __restrict__ cw,
                            const float* __restrict__ cb,
                            float* __restrict__ xm_act,
                            unsigned short* __restrict__ xm_bf, int total) {
  int i = blockIdx.x * blockDim.x + threadIdx.x;
  if (i >= total) return;
  int d = i & (DINNER - 1);
  int g = i >> 9;            // b*L + t
  int t = g & (SEQL - 1);
  float acc = cb[d];
  #pragma unroll
  for (int j = 0; j < 4; ++j) {
    int tt = t - 3 + j;
    if (tt >= 0) acc += xm_raw[(size_t)(g - 3 + j) * DINNER + d] * cw[d * 4 + j];
  }
  float s = acc / (1.0f + __expf(-acc));  // silu
  xm_act[i] = s;
  xm_bf[i]  = f32_to_bf16(s);
}

// ---------------------------------------------------------------------------
// dt = softplus(dt_raw @ dt_proj_w + dt_proj_b); dt_raw = dbc[:, :16]
// ---------------------------------------------------------------------------
__global__ __launch_bounds__(256) void k_dtproj(
    const float* __restrict__ dbc, const float* __restrict__ w,
    const float* __restrict__ bias, float* __restrict__ dt) {
  int row = blockIdx.x;  // 0..B*L-1
  __shared__ float s[DTRANK];
  if (threadIdx.x < DTRANK) s[threadIdx.x] = dbc[(size_t)row * 48 + threadIdx.x];
  __syncthreads();
  #pragma unroll
  for (int rep = 0; rep < 2; ++rep) {
    int d = threadIdx.x + rep * 256;
    float a = bias[d];
    #pragma unroll
    for (int r = 0; r < DTRANK; ++r) a += s[r] * w[r * DINNER + d];
    float sp = (a > 20.0f) ? a : __logf(1.0f + __expf(a));
    dt[(size_t)row * DINNER + d] = sp;
  }
}

// ---------------------------------------------------------------------------
// Selective scan, chunked (32 chunks x 64 steps).
// ---------------------------------------------------------------------------
__global__ __launch_bounds__(256) void k_scan_phase1(
    const float* __restrict__ dbc, const float* __restrict__ dt,
    const float* __restrict__ xm, const float* __restrict__ A_log,
    float* __restrict__ P, float* __restrict__ hloc) {
  const int tid = threadIdx.x;
  const int d = blockIdx.x * 256 + tid;
  const int c = blockIdx.y;
  const int b = blockIdx.z;
  __shared__ float lB[CHUNKL][NSTATE];
  #pragma unroll
  for (int j = 0; j < 4; ++j) {
    int i = tid + 256 * j;
    int t = i >> 4, n = i & 15;
    lB[t][n] = dbc[((size_t)(b * SEQL + c * CHUNKL + t)) * 48 + 16 + n];
  }
  __syncthreads();
  float An[NSTATE];
  #pragma unroll
  for (int n = 0; n < NSTATE; ++n) An[n] = -__expf(A_log[d * NSTATE + n]);
  float Pp[NSTATE], h[NSTATE];
  #pragma unroll
  for (int n = 0; n < NSTATE; ++n) { Pp[n] = 1.0f; h[n] = 0.0f; }
  for (int t = 0; t < CHUNKL; ++t) {
    size_t g = (size_t)(b * SEQL + c * CHUNKL + t);
    float dtv = dt[g * DINNER + d];
    float dtx = dtv * xm[g * DINNER + d];
    #pragma unroll
    for (int n = 0; n < NSTATE; ++n) {
      float dA = __expf(dtv * An[n]);
      h[n]  = dA * h[n] + dtx * lB[t][n];
      Pp[n] *= dA;
    }
  }
  size_t base = (((size_t)(b * DINNER + d)) * NCHUNK + c) * NSTATE;
  #pragma unroll
  for (int n = 0; n < NSTATE; ++n) { P[base + n] = Pp[n]; hloc[base + n] = h[n]; }
}

__global__ void k_scan_phase2(const float* __restrict__ P,
                              const float* __restrict__ hloc,
                              float* __restrict__ hinit) {
  int i = blockIdx.x * blockDim.x + threadIdx.x;  // b*DINNER + d
  float h[NSTATE];
  #pragma unroll
  for (int n = 0; n < NSTATE; ++n) h[n] = 0.0f;
  size_t base = (size_t)i * NCHUNK * NSTATE;
  for (int c = 0; c < NCHUNK; ++c) {
    size_t o = base + (size_t)c * NSTATE;
    #pragma unroll
    for (int n = 0; n < NSTATE; ++n) {
      hinit[o + n] = h[n];
      h[n] = P[o + n] * h[n] + hloc[o + n];
    }
  }
}

__global__ __launch_bounds__(256) void k_scan_phase3(
    const float* __restrict__ dbc, const float* __restrict__ dt,
    const float* __restrict__ xm, const float* __restrict__ z,
    const float* __restrict__ A_log, const float* __restrict__ D_skip,
    const float* __restrict__ hinit, unsigned short* __restrict__ ygate) {
  const int tid = threadIdx.x;
  const int d = blockIdx.x * 256 + tid;
  const int c = blockIdx.y;
  const int b = blockIdx.z;
  __shared__ float lB[CHUNKL][NSTATE];
  __shared__ float lC[CHUNKL][NSTATE];
  #pragma unroll
  for (int j = 0; j < 8; ++j) {
    int i = tid + 256 * j;
    int t = i >> 5, q = i & 31;
    float v = dbc[((size_t)(b * SEQL + c * CHUNKL + t)) * 48 + 16 + q];
    if (q < 16) lB[t][q] = v;
    else        lC[t][q - 16] = v;
  }
  __syncthreads();
  float An[NSTATE], h[NSTATE];
  size_t base = (((size_t)(b * DINNER + d)) * NCHUNK + c) * NSTATE;
  #pragma unroll
  for (int n = 0; n < NSTATE; ++n) {
    An[n] = -__expf(A_log[d * NSTATE + n]);
    h[n]  = hinit[base + n];
  }
  const float Dv = D_skip[d];
  for (int t = 0; t < CHUNKL; ++t) {
    size_t g = (size_t)(b * SEQL + c * CHUNKL + t);
    float dtv = dt[g * DINNER + d];
    float xv  = xm[g * DINNER + d];
    float dtx = dtv * xv;
    float y = 0.0f;
    #pragma unroll
    for (int n = 0; n < NSTATE; ++n) {
      float dA = __expf(dtv * An[n]);
      h[n] = dA * h[n] + dtx * lB[t][n];
      y += h[n] * lC[t][n];
    }
    float zv  = z[g * DINNER + d];
    float sil = zv / (1.0f + __expf(-zv));
    ygate[g * DINNER + d] = f32_to_bf16((y + xv * Dv) * sil);
  }
}

// ---------------------------------------------------------------------------
// im2col for the strided downsample conv (k=3,s=2,p=1): rows [B*1024], K=768
// ---------------------------------------------------------------------------
__global__ void k_im2col(const float* __restrict__ x_out,
                         unsigned short* __restrict__ col, int total) {
  int i = blockIdx.x * blockDim.x + threadIdx.x;
  if (i >= total) return;
  int cc  = i % 768;
  int row = i / 768;
  int tp  = row & 1023;
  int b   = row >> 10;
  int tap = cc / 256, ci = cc % 256;
  int t = 2 * tp - 1 + tap;
  float v = 0.0f;
  if (t >= 0 && t < SEQL) v = x_out[((size_t)(b * SEQL + t)) * DMODEL + ci];
  col[(size_t)row * 768 + cc] = f32_to_bf16(v);
}

// ---------------------------------------------------------------------------
// bias + LayerNorm over DMODEL=256 (one block per row)
// ---------------------------------------------------------------------------
__global__ __launch_bounds__(256) void k_layernorm(
    const float* __restrict__ xd_raw, const float* __restrict__ bias,
    const float* __restrict__ g, const float* __restrict__ bl,
    float* __restrict__ out) {
  int row = blockIdx.x;
  int c = threadIdx.x;
  float v = xd_raw[(size_t)row * DMODEL + c] + bias[c];
  __shared__ float red[DMODEL];
  red[c] = v;
  __syncthreads();
  for (int s = 128; s > 0; s >>= 1) {
    if (c < s) red[c] += red[c + s];
    __syncthreads();
  }
  float mu = red[0] * (1.0f / DMODEL);
  __syncthreads();
  float dv = v - mu;
  red[c] = dv * dv;
  __syncthreads();
  for (int s = 128; s > 0; s >>= 1) {
    if (c < s) red[c] += red[c + s];
    __syncthreads();
  }
  float var = red[0] * (1.0f / DMODEL);
  out[(size_t)row * DMODEL + c] = dv * rsqrtf(var + 1e-5f) * g[c] + bl[c];
}

// ---------------------------------------------------------------------------
extern "C" void kernel_launch(void* const* d_in, const int* in_sizes, int n_in,
                              void* d_out, int out_size, void* d_ws,
                              size_t ws_size, hipStream_t stream) {
  const float* x        = (const float*)d_in[0];
  const float* in_proj  = (const float*)d_in[1];
  const float* conv_w   = (const float*)d_in[2];
  const float* conv_b   = (const float*)d_in[3];
  const float* x_proj   = (const float*)d_in[4];
  const float* dt_projw = (const float*)d_in[5];
  const float* dt_projb = (const float*)d_in[6];
  const float* A_log    = (const float*)d_in[7];
  const float* D_skip   = (const float*)d_in[8];
  const float* out_proj = (const float*)d_in[9];
  const float* down_w   = (const float*)d_in[10];
  const float* down_b   = (const float*)d_in[11];
  const float* ln_g     = (const float*)d_in[12];
  const float* ln_b     = (const float*)d_in[13];

  const size_t BL   = (size_t)BATCH * SEQL;       // 32768
  const size_t BLDI = BL * DINNER;                // 16,777,216
  const size_t CHSZ = (size_t)BATCH * DINNER * NCHUNK * NSTATE;  // 4,194,304

  // d_out = [xd : B*1024*256 f32][x_mamba : B*2048*256 f32]
  float* xd_out = (float*)d_out;
  float* x_out  = xd_out + (size_t)BATCH * (SEQL / 2) * DMODEL;

  char* ws = (char*)d_ws;
  size_t o = 0;
  auto alloc = [&](size_t bytes) {
    size_t r = o;
    o += (bytes + 255) & ~(size_t)255;
    return r;
  };
  size_t off_xmraw = alloc(BLDI * 4);       // f32; later reused for ygate/im2col
  size_t off_z     = alloc(BLDI * 4);
  size_t off_xmact = alloc(BLDI * 4);
  size_t off_xmbf  = alloc(BLDI * 2);       // bf16; later reused for xd_raw f32
  size_t off_dbc   = alloc(BL * 48 * 4);
  size_t off_dt    = alloc(BLDI * 4);
  size_t off_P     = alloc(CHSZ * 4);
  size_t off_hloc  = alloc(CHSZ * 4);
  size_t off_hinit = alloc(CHSZ * 4);
  size_t off_Wi    = alloc((size_t)DMODEL * 1024 * 2);   // transposed [1024][256]
  size_t off_Wx    = alloc((size_t)DINNER * 48 * 2);     // transposed [48][512]
  size_t off_Wo    = alloc((size_t)DINNER * DMODEL * 2); // transposed [256][512]
  size_t off_Wdn   = alloc((size_t)256 * 768 * 2);       // [256][768]
  size_t off_xbf   = alloc(BL * DMODEL * 2);
  (void)ws_size; (void)n_in; (void)in_sizes; (void)out_size;

  float* xm_raw = (float*)(ws + off_xmraw);
  float* zbuf   = (float*)(ws + off_z);
  float* xm_act = (float*)(ws + off_xmact);
  unsigned short* xm_bf = (unsigned short*)(ws + off_xmbf);
  float* dbc    = (float*)(ws + off_dbc);
  float* dtb    = (float*)(ws + off_dt);
  float* Pb     = (float*)(ws + off_P);
  float* hloc   = (float*)(ws + off_hloc);
  float* hinit  = (float*)(ws + off_hinit);
  unsigned short* Wi  = (unsigned short*)(ws + off_Wi);
  unsigned short* Wx  = (unsigned short*)(ws + off_Wx);
  unsigned short* Wo  = (unsigned short*)(ws + off_Wo);
  unsigned short* Wdn = (unsigned short*)(ws + off_Wdn);
  unsigned short* xbf = (unsigned short*)(ws + off_xbf);
  // aliases (lifetimes are disjoint under stream ordering):
  unsigned short* ygate  = (unsigned short*)(ws + off_xmraw);
  unsigned short* im2col = (unsigned short*)(ws + off_xmraw);
  float* xd_raw = (float*)(ws + off_xmbf);

  // 1) weight / input conversions to bf16 (weights transposed to N x K)
  {
    int n;
    n = DMODEL * 1024;
    k_cvt_transpose<<<(n + 255) / 256, 256, 0, stream>>>(in_proj, Wi, DMODEL,
                                                         1024);
    n = DINNER * 48;
    k_cvt_transpose<<<(n + 255) / 256, 256, 0, stream>>>(x_proj, Wx, DINNER,
                                                         48);
    n = DINNER * DMODEL;
    k_cvt_transpose<<<(n + 255) / 256, 256, 0, stream>>>(out_proj, Wo, DINNER,
                                                         DMODEL);
    n = 256 * 768;
    k_wdown_cvt<<<(n + 255) / 256, 256, 0, stream>>>(down_w, Wdn);
    n = (int)(BL * DMODEL);
    k_f32_to_bf16<<<(n + 255) / 256, 256, 0, stream>>>(x, xbf, n);
  }

  // 2) in_proj GEMM [32768,256]x[256,1024], split epilogue -> xm_raw, z
  {
    dim3 grid(32768 / 128, 1024 / 64);
    k_wmma_gemm<1, false><<<grid, 256, 0, stream>>>(
        xbf, DMODEL, Wi, DMODEL, xm_raw, zbuf, 512, 32768, 1024, DMODEL);
  }

  // 3) causal depthwise conv + SiLU
  {
    int total = (int)BLDI;
    k_conv_silu<<<(total + 255) / 256, 256, 0, stream>>>(
        xm_raw, conv_w, conv_b, xm_act, xm_bf, total);
  }

  // 4) x_proj GEMM [32768,512]x[512,48] -> dbc (N=48 -> guarded)
  {
    dim3 grid(32768 / 128, 1);
    k_wmma_gemm<0, true><<<grid, 256, 0, stream>>>(
        xm_bf, DINNER, Wx, DINNER, dbc, nullptr, 48, 32768, 48, DINNER);
  }

  // 5) dt projection + softplus
  k_dtproj<<<32768, 256, 0, stream>>>(dbc, dt_projw, dt_projb, dtb);

  // 6-8) chunked selective scan
  {
    dim3 grid(DINNER / 256, NCHUNK, BATCH);
    k_scan_phase1<<<grid, 256, 0, stream>>>(dbc, dtb, xm_act, A_log, Pb, hloc);
    k_scan_phase2<<<(BATCH * DINNER) / 256, 256, 0, stream>>>(Pb, hloc, hinit);
    k_scan_phase3<<<grid, 256, 0, stream>>>(dbc, dtb, xm_act, zbuf, A_log,
                                            D_skip, hinit, ygate);
  }

  // 9) out_proj GEMM [32768,512]x[512,256] -> x_out (second tuple output)
  {
    dim3 grid(32768 / 128, DMODEL / 64);
    k_wmma_gemm<0, false><<<grid, 256, 0, stream>>>(
        ygate, DINNER, Wo, DINNER, x_out, nullptr, DMODEL, 32768, DMODEL,
        DINNER);
  }

  // 10) im2col for downsample conv
  {
    int total = BATCH * 1024 * 768;
    k_im2col<<<(total + 255) / 256, 256, 0, stream>>>(x_out, im2col, total);
  }

  // 11) downsample GEMM [16384,768]x[768,256] -> xd_raw
  {
    dim3 grid(16384 / 128, DMODEL / 64);
    k_wmma_gemm<0, false><<<grid, 256, 0, stream>>>(
        im2col, 768, Wdn, 768, xd_raw, nullptr, DMODEL, 16384, DMODEL, 768);
  }

  // 12) bias + LayerNorm -> xd_out (first tuple output)
  k_layernorm<<<BATCH * 1024, 256, 0, stream>>>(xd_raw, down_b, ln_g, ln_b,
                                                xd_out);
}
